// Transform_5506148073809
// MI455X (gfx1250) — compile-verified, matrix operand order checked by reading, not run on hardware
//
#include <hip/hip_runtime.h>
#include <math.h>

// ---------------- problem constants ----------------
constexpr int Bx = 2;
constexpr int Dm = 352;
constexpr int Hs = 80;
constexpr int Ws = 80;
constexpr int Ss = Hs * Ws;          // 6400
constexpr int nHd = 8;
constexpr int dHd = 44;
constexpr int nPt = 6;
constexpr int nLy = 12;
constexpr int Ffn = 4 * Dm;          // 1408
constexpr int Mrows = Bx * Ss;       // 12800
constexpr int NFq = 176;
constexpr int COUT = 176;
constexpr int OCH = 3 * COUT + Dm;   // 880 output channels

// ---------------- WMMA types ----------------
typedef __attribute__((ext_vector_type(16))) __bf16 v16bf;
typedef __attribute__((ext_vector_type(8)))  float  v8f;

// =====================================================================
// Tiled GEMM:  C[M,N] (+)= A[M,K](bf16) * B[K,N](f32->bf16) + bias
//  - A staged into LDS with GLOBAL_LOAD_ASYNC_TO_LDS_B128 (ASYNCcnt)
//  - B staged with f32 loads converted to bf16 (handles strided conv-B)
//  - core op: v_wmma_f32_16x16x32_bf16, f32 accumulate
//  - block 128x128, 8 waves, each wave 32x64 = 2x4 WMMA tiles (8 wmma/kstep)
//  - mode 0: store acc+bias ; mode 1: store gelu(acc+bias) ; mode 2: C += acc
//    outBf: store C as bf16 (modes 0/1 only)
//  - tap mode (imgW>0): 3x3 conv tap; shifted token row, zero-masked at edges
// =====================================================================
#define BMt 128
#define BNt 128
#define BKt 32
#define LDS_STRIDE 40                         // bf16 elements per LDS row (pad)
#define LDSA_BYTES (BMt * LDS_STRIDE * 2)     // 10240
#define LDSB_BYTES (BNt * LDS_STRIDE * 2)     // 10240
#define GEMM_LDS (LDSA_BYTES + LDSB_BYTES)    // 20480

__global__ __launch_bounds__(256)
void gemm_wmma(const __bf16* __restrict__ A, const float* __restrict__ Bm,
               const float* __restrict__ bias, void* __restrict__ Cv,
               int M, int N, int K, int lda, int ldbK, int ldbN, int ldc,
               int mode, int outBf, int tapDy, int tapDx, int imgW, int imgH)
{
  extern __shared__ char smem[];              // dynamic LDS, offset 0
  __bf16* lA = (__bf16*)smem;
  __bf16* lB = (__bf16*)(smem + LDSA_BYTES);

  const int tid  = threadIdx.x;
  const int lane = tid & 31;
  const int wave = tid >> 5;
  const int wm   = wave >> 1;        // 0..3  row group of 32
  const int wn   = wave & 1;         // 0..1  col group of 64
  const int l16  = lane & 15;
  const int khalf = lane >> 4;
  const int blockRow = blockIdx.y * BMt;
  const int blockCol = blockIdx.x * BNt;
  const bool tap = (imgW > 0);
  const int Simg = imgW * imgH;

  v8f acc[2][4] = {};

  for (int k0 = 0; k0 < K; k0 += BKt) {
    // ---- stage A tile: 128 rows x 32 bf16 = 512 x 16B chunks, async ----
#pragma unroll
    for (int j = 0; j < 2; ++j) {
      int i = tid + j * 256;                  // 0..511
      int m = i >> 2;                         // row in tile
      int c = i & 3;                          // 16B chunk in row
      int g = blockRow + m;
      bool valid = (g < M);
      int srow = g;
      if (tap && valid) {
        int p  = g % Simg;
        int py = p / imgW, px = p % imgW;
        int qx = px + tapDx, qy = py + tapDy;
        valid = (qx >= 0) && (qx < imgW) && (qy >= 0) && (qy < imgH);
        srow  = g + tapDy * imgW + tapDx;
      }
      unsigned lofs = (unsigned)(m * LDS_STRIDE * 2 + c * 16);  // byte offset
      if (valid) {
        const __bf16* gp = A + (size_t)srow * lda + k0 + c * 8;
        asm volatile("global_load_async_to_lds_b128 %0, %1, off"
                     :: "v"(lofs), "v"(gp) : "memory");
      } else {
        uint4 z = make_uint4(0u, 0u, 0u, 0u);
        *(uint4*)(smem + lofs) = z;           // ds_store_b128 zero fill
      }
    }
    // ---- stage B tile: 32 x 128 f32 -> bf16 (transposed, n-major) ----
#pragma unroll
    for (int j = 0; j < 16; ++j) {
      int i = tid + j * 256;                  // 0..4095
      int n = i & 127;
      int k = i >> 7;                         // 0..31
      int col = blockCol + n;
      float v = 0.f;
      if (col < N && (k0 + k) < K)
        v = Bm[(size_t)(k0 + k) * ldbK + (size_t)col * ldbN];
      lB[n * LDS_STRIDE + k] = (__bf16)v;
    }
    // prefetch next A tile
    if (k0 + BKt < K) {
      int g = blockRow + (tid >> 1);
      if (g < M)
        __builtin_prefetch(A + (size_t)g * lda + k0 + BKt, 0, 1);
    }
    asm volatile("s_wait_asynccnt 0x0" ::: "memory");
    __syncthreads();

    // ---- WMMA: wave computes 32x64 via 2x4 tiles of 16x16x32 ----
    union Fr { uint4 u[2]; v16bf v; };
    Fr afr[2], bfr[4];
#pragma unroll
    for (int mt = 0; mt < 2; ++mt) {
      const __bf16* p = lA + (wm * 32 + mt * 16 + l16) * LDS_STRIDE + khalf * 8;
      afr[mt].u[0] = *(const uint4*)(p);
      afr[mt].u[1] = *(const uint4*)(p + 16);
    }
#pragma unroll
    for (int nt = 0; nt < 4; ++nt) {
      const __bf16* p = lB + (wn * 64 + nt * 16 + l16) * LDS_STRIDE + khalf * 8;
      bfr[nt].u[0] = *(const uint4*)(p);
      bfr[nt].u[1] = *(const uint4*)(p + 16);
    }
#pragma unroll
    for (int mt = 0; mt < 2; ++mt)
#pragma unroll
      for (int nt = 0; nt < 4; ++nt)
        acc[mt][nt] = __builtin_amdgcn_wmma_f32_16x16x32_bf16(
            false, afr[mt].v, false, bfr[nt].v, (short)0, acc[mt][nt],
            false, false);
    __syncthreads();
  }

  // ---- epilogue ----
  float* Cf = (float*)Cv;
  __bf16* Ch = (__bf16*)Cv;
#pragma unroll
  for (int mt = 0; mt < 2; ++mt) {
#pragma unroll
    for (int nt = 0; nt < 4; ++nt) {
      int rbase = blockRow + wm * 32 + mt * 16 + khalf * 8;
      int col   = blockCol + wn * 64 + nt * 16 + l16;
      if (col >= N) continue;
      float bv = bias ? bias[col] : 0.f;
#pragma unroll
      for (int r = 0; r < 8; ++r) {
        int row = rbase + r;
        if (row >= M) continue;
        float x = acc[mt][nt][r] + bv;
        if (mode == 1)       // exact GELU
          x = 0.5f * x * (1.f + erff(x * 0.70710678118654752f));
        size_t ci = (size_t)row * ldc + col;
        if (mode == 2)       Cf[ci] += x;
        else if (outBf)      Ch[ci]  = (__bf16)x;
        else                 Cf[ci]  = x;
      }
    }
  }
}

// =====================================================================
// Sine positional embedding (+ level embed)   pos[S, D]
// =====================================================================
__global__ void pos_kernel(float* __restrict__ pos,
                           const float* __restrict__ level_embed)
{
  int idx = blockIdx.x * blockDim.x + threadIdx.x;
  if (idx >= Ss * Dm) return;
  int d = idx % Dm;
  int s = idx / Dm;
  int row = s / Ws, col = s % Ws;
  const float scale = 6.283185307179586f;
  float e;
  int j;
  if (d < NFq) { e = (row + 1) / (Hs + 1e-6f) * scale; j = d; }
  else         { e = (col + 1) / (Ws + 1e-6f) * scale; j = d - NFq; }
  float jj = (float)(2 * (j >> 1)) / (float)NFq;
  float dt = powf(10000.f, jj);
  float v = e / dt;
  float p = (j & 1) ? cosf(v) : sinf(v);
  pos[idx] = p + level_embed[d];
}

// x [B, D, S] (NCHW) -> src f32 [B,S,D] and src16 bf16 [B,S,D]
__global__ void transpose_kernel(const float* __restrict__ x,
                                 float* __restrict__ src,
                                 __bf16* __restrict__ src16)
{
  int idx = blockIdx.x * blockDim.x + threadIdx.x;
  if (idx >= Bx * Ss * Dm) return;
  int d = idx % Dm;
  int t = idx / Dm;
  int s = t % Ss;
  int b = t / Ss;
  float v = x[((size_t)b * Dm + d) * Ss + s];
  src[idx] = v;
  src16[idx] = (__bf16)v;
}

// q16 = bf16(src + pos), batch-broadcast pos
__global__ void addpos_kernel(const float* __restrict__ src,
                              const float* __restrict__ pos,
                              __bf16* __restrict__ q16)
{
  int idx = blockIdx.x * blockDim.x + threadIdx.x;
  if (idx >= Bx * Ss * Dm) return;
  q16[idx] = (__bf16)(src[idx] + pos[idx % (Ss * Dm)]);
}

// softmax over NP=6 per (b,s,h), in place
__global__ void softmax6_kernel(float* __restrict__ a)
{
  int i = blockIdx.x * blockDim.x + threadIdx.x;
  if (i >= Bx * Ss * nHd) return;
  float* p = a + (size_t)i * nPt;
  float m = p[0];
#pragma unroll
  for (int j = 1; j < nPt; ++j) m = fmaxf(m, p[j]);
  float e[nPt], s = 0.f;
#pragma unroll
  for (int j = 0; j < nPt; ++j) { e[j] = expf(p[j] - m); s += e[j]; }
  float inv = 1.f / s;
#pragma unroll
  for (int j = 0; j < nPt; ++j) p[j] = e[j] * inv;
}

// MSDA bilinear sampling: one thread per (b,s,h,d) -> bf16 samp
__global__ void msda_kernel(const float* __restrict__ val,
                            const float* __restrict__ off,
                            const float* __restrict__ attn,
                            __bf16* __restrict__ samp16)
{
  int idx = blockIdx.x * blockDim.x + threadIdx.x;
  if (idx >= Bx * Ss * nHd * dHd) return;
  int d  = idx % dHd;
  int t  = idx / dHd;
  int h  = t % nHd;
  int bs = t / nHd;          // b*S + s
  int s  = bs % Ss;
  int b  = bs / Ss;
  int py = s / Ws, px = s % Ws;
  float refx = (px + 0.5f) / Ws;
  float refy = (py + 0.5f) / Hs;
  const float* ob = off  + (size_t)bs * (nHd * nPt * 2) + h * (nPt * 2);
  const float* ab = attn + (size_t)bs * (nHd * nPt)     + h * nPt;
  const float* vb = val  + (size_t)b * Ss * Dm + h * dHd + d;
  float out = 0.f;
#pragma unroll
  for (int p = 0; p < nPt; ++p) {
    float lx = refx + ob[2 * p]     / (float)Ws;
    float ly = refy + ob[2 * p + 1] / (float)Hs;
    float x = lx * Ws - 0.5f;
    float y = ly * Hs - 0.5f;
    float x0f = floorf(x), y0f = floorf(y);
    int x0 = (int)x0f, y0 = (int)y0f;
    float dx = x - x0f, dy = y - y0f;
    float w00 = (1.f - dx) * (1.f - dy);
    float w10 = dx * (1.f - dy);
    float w01 = (1.f - dx) * dy;
    float w11 = dx * dy;
    float acc = 0.f;
    if (x0 >= 0     && x0 < Ws     && y0 >= 0     && y0 < Hs)
      acc += vb[(size_t)(y0 * Ws + x0) * Dm] * w00;
    if (x0 + 1 >= 0 && x0 + 1 < Ws && y0 >= 0     && y0 < Hs)
      acc += vb[(size_t)(y0 * Ws + x0 + 1) * Dm] * w10;
    if (x0 >= 0     && x0 < Ws     && y0 + 1 >= 0 && y0 + 1 < Hs)
      acc += vb[(size_t)((y0 + 1) * Ws + x0) * Dm] * w01;
    if (x0 + 1 >= 0 && x0 + 1 < Ws && y0 + 1 >= 0 && y0 + 1 < Hs)
      acc += vb[(size_t)((y0 + 1) * Ws + x0 + 1) * Dm] * w11;
    out += ab[p] * acc;
  }
  samp16[(size_t)bs * Dm + h * dHd + d] = (__bf16)out;
}

// src = LN(src + tmp); also writes bf16 src16 and optional bf16 save16
// one wave32 per row, D = 352 = 11*32
__global__ __launch_bounds__(256)
void add_ln_kernel(float* __restrict__ src, const float* __restrict__ tmp,
                   const float* __restrict__ g, const float* __restrict__ be,
                   __bf16* __restrict__ src16, __bf16* __restrict__ save16,
                   int rows)
{
  int wave = threadIdx.x >> 5;
  int lane = threadIdx.x & 31;
  int row = blockIdx.x * 8 + wave;
  if (row >= rows) return;
  float* sr = src + (size_t)row * Dm;
  const float* tr = tmp + (size_t)row * Dm;
  float y[11];
  float s = 0.f;
#pragma unroll
  for (int i = 0; i < 11; ++i) {
    int d = lane + i * 32;
    y[i] = sr[d] + tr[d];
    s += y[i];
  }
#pragma unroll
  for (int o = 16; o > 0; o >>= 1) s += __shfl_xor(s, o, 32);
  float mean = s / (float)Dm;
  float v = 0.f;
#pragma unroll
  for (int i = 0; i < 11; ++i) { float t = y[i] - mean; v += t * t; }
#pragma unroll
  for (int o = 16; o > 0; o >>= 1) v += __shfl_xor(v, o, 32);
  float rs = rsqrtf(v / (float)Dm + 1e-5f);
#pragma unroll
  for (int i = 0; i < 11; ++i) {
    int d = lane + i * 32;
    float o = (y[i] - mean) * rs * g[d] + be[d];
    sr[d] = o;
    src16[(size_t)row * Dm + d] = (__bf16)o;
    if (save16) save16[(size_t)row * Dm + d] = (__bf16)o;
  }
}

// conv accum [B,S,176] -> +bias, BN, ReLU -> out NCHW at channel offset
__global__ void bnrelu_out_kernel(const float* __restrict__ acc,
                                  const float* __restrict__ cb,
                                  const float* __restrict__ g,
                                  const float* __restrict__ beta,
                                  const float* __restrict__ m,
                                  const float* __restrict__ v,
                                  float* __restrict__ out, int chOff)
{
  int idx = blockIdx.x * blockDim.x + threadIdx.x;
  if (idx >= Bx * COUT * Ss) return;
  int p  = idx % Ss;
  int t  = idx / Ss;
  int co = t % COUT;
  int b  = t / COUT;
  float x = acc[((size_t)b * Ss + p) * COUT + co] + cb[co];
  x = (x - m[co]) * rsqrtf(v[co] + 1e-5f) * g[co] + beta[co];
  out[((size_t)b * OCH + chOff + co) * Ss + p] = fmaxf(x, 0.f);
}

// passthrough of x into output channels [528, 880)
__global__ void copyx_kernel(const float* __restrict__ x,
                             float* __restrict__ out)
{
  int idx = blockIdx.x * blockDim.x + threadIdx.x;
  if (idx >= Bx * Dm * Ss) return;
  int p = idx % Ss;
  int t = idx / Ss;
  int c = t % Dm;
  int b = t / Dm;
  out[((size_t)b * OCH + 3 * COUT + c) * Ss + p] = x[((size_t)b * Dm + c) * Ss + p];
}

// =====================================================================
// host orchestration
// =====================================================================
static inline int ceil_div(int a, int b) { return (a + b - 1) / b; }

extern "C" void kernel_launch(void* const* d_in, const int* in_sizes, int n_in,
                              void* d_out, int out_size, void* d_ws, size_t ws_size,
                              hipStream_t stream)
{
  const float* x          = (const float*)d_in[0];
  const float* level_emb  = (const float*)d_in[1];
  const float* W_off      = (const float*)d_in[2];
  const float* b_off      = (const float*)d_in[3];
  const float* W_attn     = (const float*)d_in[4];
  const float* b_attn     = (const float*)d_in[5];
  const float* W_val      = (const float*)d_in[6];
  const float* b_val      = (const float*)d_in[7];
  const float* W_out      = (const float*)d_in[8];
  const float* b_out      = (const float*)d_in[9];
  const float* ln1_g      = (const float*)d_in[10];
  const float* ln1_b      = (const float*)d_in[11];
  const float* ln2_g      = (const float*)d_in[12];
  const float* ln2_b      = (const float*)d_in[13];
  const float* W_ff1      = (const float*)d_in[14];
  const float* b_ff1      = (const float*)d_in[15];
  const float* W_ff2      = (const float*)d_in[16];
  const float* b_ff2      = (const float*)d_in[17];
  const float* conv_w[3]  = { (const float*)d_in[18], (const float*)d_in[20], (const float*)d_in[22] };
  const float* conv_b[3]  = { (const float*)d_in[19], (const float*)d_in[21], (const float*)d_in[23] };
  const float* bn_g[3]    = { (const float*)d_in[24], (const float*)d_in[28], (const float*)d_in[32] };
  const float* bn_b[3]    = { (const float*)d_in[25], (const float*)d_in[29], (const float*)d_in[33] };
  const float* bn_m[3]    = { (const float*)d_in[26], (const float*)d_in[30], (const float*)d_in[34] };
  const float* bn_v[3]    = { (const float*)d_in[27], (const float*)d_in[31], (const float*)d_in[35] };
  float* out = (float*)d_out;

  // workspace carve-up (byte pointer; all chunks 16B aligned)
  char* wsb = (char*)d_ws;
  auto allocF = [&](size_t n) { float* p = (float*)wsb;  wsb += n * sizeof(float);  return p; };
  auto allocH = [&](size_t n) { __bf16* p = (__bf16*)wsb; wsb += n * sizeof(__bf16); return p; };

  float*  pos    = allocF((size_t)Ss * Dm);
  float*  src    = allocF((size_t)Mrows * Dm);
  __bf16* src16  = allocH((size_t)Mrows * Dm);
  __bf16* q16    = allocH((size_t)Mrows * Dm);
  float*  offb   = allocF((size_t)Mrows * nHd * nPt * 2);
  float*  attn   = allocF((size_t)Mrows * nHd * nPt);
  float*  valb   = allocF((size_t)Mrows * Dm);
  __bf16* samp16 = allocH((size_t)Mrows * Dm);
  float*  tmpb   = allocF((size_t)Mrows * Dm);
  __bf16* ff16   = allocH((size_t)Mrows * Ffn);
  __bf16* save16[3];
  save16[0] = allocH((size_t)Mrows * Dm);
  save16[1] = allocH((size_t)Mrows * Dm);
  save16[2] = allocH((size_t)Mrows * Dm);
  float*  cacc   = allocF((size_t)Mrows * COUT);

  const int TB = 256;
  auto gemm = [&](const __bf16* Ap, const float* Bp, const float* biasp, void* Cp,
                  int M_, int N_, int K_, int lda_, int ldbK_, int ldbN_, int ldc_,
                  int mode, int outBf, int dy = 0, int dx = 0, int iw = 0, int ih = 0) {
    dim3 grid(ceil_div(N_, BNt), ceil_div(M_, BMt));
    gemm_wmma<<<grid, dim3(256), GEMM_LDS, stream>>>(Ap, Bp, biasp, Cp, M_, N_, K_,
                                                     lda_, ldbK_, ldbN_, ldc_,
                                                     mode, outBf, dy, dx, iw, ih);
  };

  pos_kernel<<<ceil_div(Ss * Dm, TB), TB, 0, stream>>>(pos, level_emb);
  transpose_kernel<<<ceil_div(Mrows * Dm, TB), TB, 0, stream>>>(x, src, src16);

  for (int l = 0; l < nLy; ++l) {
    addpos_kernel<<<ceil_div(Mrows * Dm, TB), TB, 0, stream>>>(src, pos, q16);

    gemm(q16, W_off + (size_t)l * Dm * (nHd * nPt * 2), b_off + (size_t)l * (nHd * nPt * 2),
         offb, Mrows, nHd * nPt * 2, Dm, Dm, nHd * nPt * 2, 1, nHd * nPt * 2, 0, 0);
    gemm(q16, W_attn + (size_t)l * Dm * (nHd * nPt), b_attn + (size_t)l * (nHd * nPt),
         attn, Mrows, nHd * nPt, Dm, Dm, nHd * nPt, 1, nHd * nPt, 0, 0);
    softmax6_kernel<<<ceil_div(Mrows * nHd, TB), TB, 0, stream>>>(attn);

    gemm(src16, W_val + (size_t)l * Dm * Dm, b_val + (size_t)l * Dm,
         valb, Mrows, Dm, Dm, Dm, Dm, 1, Dm, 0, 0);
    msda_kernel<<<ceil_div(Mrows * nHd * dHd, TB), TB, 0, stream>>>(valb, offb, attn, samp16);

    gemm(samp16, W_out + (size_t)l * Dm * Dm, b_out + (size_t)l * Dm,
         tmpb, Mrows, Dm, Dm, Dm, Dm, 1, Dm, 0, 0);
    add_ln_kernel<<<ceil_div(Mrows, 8), 256, 0, stream>>>(
        src, tmpb, ln1_g + (size_t)l * Dm, ln1_b + (size_t)l * Dm, src16, nullptr, Mrows);

    gemm(src16, W_ff1 + (size_t)l * Dm * Ffn, b_ff1 + (size_t)l * Ffn,
         ff16, Mrows, Ffn, Dm, Dm, Ffn, 1, Ffn, 1 /*gelu*/, 1 /*bf16 out*/);
    gemm(ff16, W_ff2 + (size_t)l * Ffn * Dm, b_ff2 + (size_t)l * Dm,
         tmpb, Mrows, Dm, Ffn, Ffn, Dm, 1, Dm, 0, 0);
    __bf16* sv = (l == 3) ? save16[0] : (l == 7) ? save16[1] : (l == 11) ? save16[2] : nullptr;
    add_ln_kernel<<<ceil_div(Mrows, 8), 256, 0, stream>>>(
        src, tmpb, ln2_g + (size_t)l * Dm, ln2_b + (size_t)l * Dm, src16, sv, Mrows);
  }

  // three 3x3 convs as 9 accumulating WMMA GEMM taps each
  for (int c = 0; c < 3; ++c) {
    int tap = 0;
    for (int dy = -1; dy <= 1; ++dy) {
      for (int dx = -1; dx <= 1; ++dx, ++tap) {
        // B element (k, co) lives at conv_w[co*352*9 + k*9 + (dy+1)*3+(dx+1)]
        gemm(save16[c], conv_w[c] + (dy + 1) * 3 + (dx + 1), nullptr, cacc,
             Mrows, COUT, Dm, Dm, /*ldbK*/ 9, /*ldbN*/ Dm * 9, COUT,
             tap == 0 ? 0 : 2, 0, dy, dx, Ws, Hs);
      }
    }
    bnrelu_out_kernel<<<ceil_div(Bx * COUT * Ss, TB), TB, 0, stream>>>(
        cacc, conv_b[c], bn_g[c], bn_b[c], bn_m[c], bn_v[c], out, c * COUT);
  }
  copyx_kernel<<<ceil_div(Bx * Dm * Ss, TB), TB, 0, stream>>>(x, out);

  (void)in_sizes; (void)n_in; (void)out_size; (void)ws_size;
}
